// TriangulationMetric_32933809226504
// MI455X (gfx1250) — compile-verified, matrix operand order checked by reading, not run on hardware
//
#include <hip/hip_runtime.h>
#include <hip/hip_bf16.h>

#define N_TRACKS 2048
#define S_FRAMES 8
#define IMG_H 224
#define IMG_W 224
#define HW (IMG_H * IMG_W)
#define NSEG 8
#define SEG_COLS (HW / NSEG)          // 6272, multiple of 16

typedef float v2f  __attribute__((ext_vector_type(2)));
typedef float v8f  __attribute__((ext_vector_type(8)));
typedef _Float16 v16h __attribute__((ext_vector_type(16)));

// ---------------------------------------------------------------------------
// Kernel 1: radix-select the 2048-th largest conf[0] value (single workgroup).
// conf in [1,5) -> positive floats -> IEEE bit pattern is order preserving.
// scal[0]=threshold bits, scal[1]=count strictly greater, scal[2..3]=counters.
// ---------------------------------------------------------------------------
__global__ __launch_bounds__(1024)
void topk_threshold_kernel(const float* __restrict__ conf0, unsigned int* __restrict__ scal) {
    __shared__ unsigned int hist[256];
    __shared__ unsigned int s_prefix, s_cntAbove, s_need;
    const int tid = threadIdx.x;
    if (tid == 0) { s_prefix = 0u; s_cntAbove = 0u; s_need = N_TRACKS; scal[2] = 0u; scal[3] = 0u; }
    __syncthreads();

    for (int pass = 3; pass >= 0; --pass) {
        if (tid < 256) hist[tid] = 0u;
        __syncthreads();
        const unsigned int shift = (unsigned int)pass * 8u;
        const unsigned int prefMask = (pass == 3) ? 0u : (0xFFFFFFFFu << (shift + 8u));
        const unsigned int prefix = s_prefix;
        for (int i = tid; i < HW; i += 1024) {
            unsigned int u = __float_as_uint(conf0[i]);
            if ((u & prefMask) == (prefix & prefMask))
                atomicAdd(&hist[(u >> shift) & 0xFFu], 1u);
        }
        __syncthreads();
        if (tid == 0) {
            unsigned int need = s_need, cum = 0u;
            int b = 255;
            for (; b > 0; --b) {
                if (cum + hist[b] >= need) break;
                cum += hist[b];
            }
            s_prefix   = prefix | ((unsigned int)b << shift);
            s_cntAbove += cum;          // strictly greater via higher bins
            s_need     = need - cum;
        }
        __syncthreads();
    }
    if (tid == 0) { scal[0] = s_prefix; scal[1] = s_cntAbove; }
}

// ---------------------------------------------------------------------------
// Kernel 2: collect indices with conf > T (slots [0,cntAbove)) and == T
// (slots [cntAbove, 2048)). Set is deterministic; order fixed by the sort.
// ---------------------------------------------------------------------------
__global__ void topk_collect_kernel(const float* __restrict__ conf0,
                                    unsigned int* __restrict__ scal,
                                    int* __restrict__ idx0) {
    const unsigned int T = scal[0], cntAbove = scal[1];
    int i = blockIdx.x * blockDim.x + threadIdx.x;
    if (i >= HW) return;
    unsigned int u = __float_as_uint(conf0[i]);
    if (u > T) {
        unsigned int pos = atomicAdd(&scal[2], 1u);
        if (pos < N_TRACKS) idx0[pos] = i;
    } else if (u == T) {
        unsigned int pos = cntAbove + atomicAdd(&scal[3], 1u);
        if (pos < N_TRACKS) idx0[pos] = i;
    }
}

// ---------------------------------------------------------------------------
// Kernel 3: bitonic sort (conf desc, idx asc) in LDS, then gather ref points
// (x,y,z,|p|^2). Single workgroup, 16 KB LDS.
// ---------------------------------------------------------------------------
__global__ __launch_bounds__(1024)
void sort_gather_kernel(const float* __restrict__ conf0, const float* __restrict__ pts0,
                        int* __restrict__ idx0, float4* __restrict__ ref4) {
    __shared__ unsigned long long key[N_TRACKS];
    const int tid = threadIdx.x;
    for (int i = tid; i < N_TRACKS; i += 1024) {
        unsigned int idx = (unsigned int)idx0[i];
        unsigned int cb  = __float_as_uint(conf0[idx]);
        key[i] = ((unsigned long long)cb << 32) | (unsigned long long)(0xFFFFFFFFu - idx);
    }
    __syncthreads();
    for (int k = 2; k <= N_TRACKS; k <<= 1) {
        for (int j = k >> 1; j > 0; j >>= 1) {
            for (int i = tid; i < N_TRACKS; i += 1024) {
                int ixj = i ^ j;
                if (ixj > i) {
                    unsigned long long a = key[i], b = key[ixj];
                    bool descSeg = ((i & k) == 0);
                    bool doSwap = descSeg ? (a < b) : (a > b);
                    if (doSwap) { key[i] = b; key[ixj] = a; }
                }
            }
            __syncthreads();
        }
    }
    for (int i = tid; i < N_TRACKS; i += 1024) {
        unsigned int idx = 0xFFFFFFFFu - (unsigned int)(key[i] & 0xFFFFFFFFull);
        idx0[i] = (int)idx;
        float x = pts0[idx * 3 + 0];
        float y = pts0[idx * 3 + 1];
        float z = pts0[idx * 3 + 2];
        ref4[i] = make_float4(x, y, z, x * x + y * y + z * z);
    }
}

// ---------------------------------------------------------------------------
// Kernel 3b: initialize argmin keys to +inf (must run every launch).
// ---------------------------------------------------------------------------
__global__ void init_keys_kernel(unsigned long long* __restrict__ keys, int n) {
    int i = blockIdx.x * blockDim.x + threadIdx.x;
    if (i < n) keys[i] = 0xFFFFFFFFFFFFFFFFull;
}

// ---------------------------------------------------------------------------
// Kernel 4: WMMA nearest-neighbor matcher, column-segmented for occupancy.
// One wave per (16 ref rows, frame, column segment); segment-local argmin is
// merged globally with GLOBAL_ATOMIC_MIN_U64 on key = (dist_bits<<32)|idx,
// which reproduces argmin's smallest-index tie-break for non-negative dists.
// Inner loop compares the row-constant-free surrogate d' = |p|^2 - 2 r.p;
// the |r|^2 offset is re-added once per wave before the atomic merge.
// grid = (128, 7, 8); block = 32 (one wave, EXEC all ones as WMMA requires).
// ---------------------------------------------------------------------------
__global__ __launch_bounds__(32)
void match_kernel(const float* __restrict__ pts, const float4* __restrict__ ref4,
                  unsigned long long* __restrict__ keys) {
    const int s       = blockIdx.y + 1;      // frames 1..7
    const int rowBase = blockIdx.x * 16;
    const int seg     = blockIdx.z;
    const int colBase = seg * SEG_COLS;
    const int colEnd  = colBase + SEG_COLS;
    const int lane    = threadIdx.x;
    const int half    = lane >> 4;           // 0: lanes 0-15, 1: lanes 16-31
    const int l15     = lane & 15;

    // A tile (16x4 f32): VGPR0 = K0 | K2, VGPR1 = K1 | K3(pad), M = lane&15.
    float4 r = ref4[rowBase + l15];
#if __has_builtin(__builtin_amdgcn_wmma_f32_16x16x4_f32)
    v2f a;
    a.x = half ? r.z : r.x;
    a.y = half ? 0.0f : r.y;
#else
    v16h a16 = {};
    if (!half) { a16[0] = (_Float16)r.x; a16[1] = (_Float16)r.y; a16[2] = (_Float16)r.z; }
#endif

    // |ref|^2 for the 8 D rows this lane holds (row = j + 8*half); only used
    // at the end to restore the true distance before the atomic merge.
    float rsq[8];
#pragma unroll
    for (int j = 0; j < 8; ++j) rsq[j] = ref4[rowBase + j + half * 8].w;

    const float* fp = pts + (size_t)s * HW * 3;
    float best[8];
    int   bidx[8];
#pragma unroll
    for (int j = 0; j < 8; ++j) { best[j] = 3.4e38f; bidx[j] = 0; }

    auto chunk = [&](int c, bool doPrefetch) {
        const int n = c + l15;
        float x = fp[n * 3 + 0];
        float y = fp[n * 3 + 1];
        float z = fp[n * 3 + 2];
        if (doPrefetch) __builtin_prefetch(fp + (size_t)(c + 16) * 3, 0, 1);
        float psq = x * x + y * y + z * z;

        v8f acc = {};
#if __has_builtin(__builtin_amdgcn_wmma_f32_16x16x4_f32)
        v2f b;
        b.x = half ? z : x;                  // K0 | K2
        b.y = half ? 0.0f : y;               // K1 | K3(pad)
        acc = __builtin_amdgcn_wmma_f32_16x16x4_f32(
            false, a, false, b, (short)0, acc, false, false);
#else
        v16h b16 = {};
        if (!half) { b16[0] = (_Float16)x; b16[1] = (_Float16)y; b16[2] = (_Float16)z; }
        acc = __builtin_amdgcn_wmma_f32_16x16x32_f16(
            false, a16, false, b16, (short)0, acc, false, false);
#endif

#pragma unroll
        for (int j = 0; j < 8; ++j) {
            // Row-constant |r|^2 dropped from the comparison: one FMA per row.
            float d = fmaf(-2.0f, acc[j], psq);
            if (d < best[j]) { best[j] = d; bidx[j] = n; }
        }
    };

    // Steady state with unconditional prefetch, then one epilogue chunk.
    for (int c = colBase; c + 16 < colEnd; c += 16) chunk(c, true);
    chunk(colEnd - 16, false);

    // Min/argmin reduction over the 16 column lanes of each half-wave.
#pragma unroll
    for (int off = 1; off < 16; off <<= 1) {
#pragma unroll
        for (int j = 0; j < 8; ++j) {
            float ob = __shfl_xor(best[j], off, 32);
            int   oi = __shfl_xor(bidx[j], off, 32);
            if (ob < best[j] || (ob == best[j] && oi < bidx[j])) { best[j] = ob; bidx[j] = oi; }
        }
    }
    if (l15 == 0) {
#pragma unroll
        for (int j = 0; j < 8; ++j) {
            float dtrue = best[j] + rsq[j];          // restore true distance
            float dcl   = fmaxf(dtrue, 0.0f);        // keep bit pattern order-preserving
            unsigned long long k =
                ((unsigned long long)__float_as_uint(dcl) << 32) |
                (unsigned long long)(unsigned int)bidx[j];
            atomicMin(&keys[(size_t)(s - 1) * N_TRACKS + rowBase + j + half * 8], k);
        }
    }
}

// ---------------------------------------------------------------------------
// Kernel 5: per-track DLT (4x4 AtA), Jacobi smallest-eigenvector, reprojection
// statistics. One thread per track. Matches: frame 0 from sorted idx0, frames
// 1..7 from the low 32 bits of the argmin keys.
// ---------------------------------------------------------------------------
__global__ __launch_bounds__(256)
void solve_kernel(const float* __restrict__ conf, const float* __restrict__ intr,
                  const float* __restrict__ w2cs, const int* __restrict__ idx0,
                  const unsigned long long* __restrict__ keys,
                  float* __restrict__ out) {
    const int t = blockIdx.x * blockDim.x + threadIdx.x;
    if (t >= N_TRACKS) return;

    float A00 = 0.f, A01 = 0.f, A02 = 0.f, A03 = 0.f;
    float A11 = 0.f, A12 = 0.f, A13 = 0.f;
    float A22 = 0.f, A23 = 0.f, A33 = 0.f;
    float tnx[S_FRAMES], tny[S_FRAMES], tcf[S_FRAMES];
    float xs0 = 0.f, ys0 = 0.f;

    for (int s = 0; s < S_FRAMES; ++s) {
        int m = (s == 0) ? idx0[t]
                         : (int)(unsigned int)(keys[(size_t)(s - 1) * N_TRACKS + t] & 0xFFFFFFFFull);
        float xs = (float)(m % IMG_W);
        float ys = (float)(m / IMG_W);
        if (s == 0) { xs0 = xs; ys0 = ys; }
        float cf = conf[(size_t)s * HW + m];
        tcf[s] = cf;
        float fx  = intr[s * 9 + 0], fy  = intr[s * 9 + 4];
        float ppx = intr[s * 9 + 2], ppy = intr[s * 9 + 5];
        float tx = (xs - ppx) / fx, ty = (ys - ppy) / fy;
        tnx[s] = tx; tny[s] = ty;
        float w = cf - 3.0f + 0.5f;
        const float* P = w2cs + s * 16;     // rows of 4x4 w2c, use rows 0..2
        float r1[4], r2[4];
#pragma unroll
        for (int k = 0; k < 4; ++k) {
            r1[k] = (tx * P[8 + k] - P[0 + k]) * w;
            r2[k] = (ty * P[8 + k] - P[4 + k]) * w;
        }
        A00 += r1[0] * r1[0] + r2[0] * r2[0];
        A01 += r1[0] * r1[1] + r2[0] * r2[1];
        A02 += r1[0] * r1[2] + r2[0] * r2[2];
        A03 += r1[0] * r1[3] + r2[0] * r2[3];
        A11 += r1[1] * r1[1] + r2[1] * r2[1];
        A12 += r1[1] * r1[2] + r2[1] * r2[2];
        A13 += r1[1] * r1[3] + r2[1] * r2[3];
        A22 += r1[2] * r1[2] + r2[2] * r2[2];
        A23 += r1[2] * r1[3] + r2[2] * r2[3];
        A33 += r1[3] * r1[3] + r2[3] * r2[3];
    }

    // Cyclic Jacobi on 4x4 symmetric matrix; constant indices -> stays in VGPRs.
    float M00 = A00, M01 = A01, M02 = A02, M03 = A03;
    float M11 = A11, M12 = A12, M13 = A13, M22 = A22, M23 = A23, M33 = A33;
    float V00 = 1.f, V01 = 0.f, V02 = 0.f, V03 = 0.f;
    float V10 = 0.f, V11 = 1.f, V12 = 0.f, V13 = 0.f;
    float V20 = 0.f, V21 = 0.f, V22 = 1.f, V23 = 0.f;
    float V30 = 0.f, V31 = 0.f, V32 = 0.f, V33 = 1.f;

#define JROT(app, aqq, apq, ar1p, ar1q, ar2p, ar2q, vp0, vq0, vp1, vq1, vp2, vq2, vp3, vq3)       \
    do {                                                                                          \
        float _apq = (apq);                                                                       \
        if (fabsf(_apq) > 1e-20f) {                                                               \
            float _th = 0.5f * ((aqq) - (app)) / _apq;                                            \
            float _tt = copysignf(1.0f, _th) / (fabsf(_th) + sqrtf(_th * _th + 1.0f));            \
            float _c = rsqrtf(_tt * _tt + 1.0f);                                                  \
            float _s = _tt * _c;                                                                  \
            float _app = (app), _aqq = (aqq);                                                     \
            (app) = _app - _tt * _apq;                                                            \
            (aqq) = _aqq + _tt * _apq;                                                            \
            (apq) = 0.0f;                                                                         \
            float _t1p = (ar1p), _t1q = (ar1q);                                                   \
            (ar1p) = _c * _t1p - _s * _t1q; (ar1q) = _s * _t1p + _c * _t1q;                       \
            float _t2p = (ar2p), _t2q = (ar2q);                                                   \
            (ar2p) = _c * _t2p - _s * _t2q; (ar2q) = _s * _t2p + _c * _t2q;                       \
            float _v;                                                                             \
            _v = (vp0); (vp0) = _c * _v - _s * (vq0); (vq0) = _s * _v + _c * (vq0);               \
            _v = (vp1); (vp1) = _c * _v - _s * (vq1); (vq1) = _s * _v + _c * (vq1);               \
            _v = (vp2); (vp2) = _c * _v - _s * (vq2); (vq2) = _s * _v + _c * (vq2);               \
            _v = (vp3); (vp3) = _c * _v - _s * (vq3); (vq3) = _s * _v + _c * (vq3);               \
        }                                                                                         \
    } while (0)

    for (int sweep = 0; sweep < 10; ++sweep) {
        JROT(M00, M11, M01, M02, M12, M03, M13, V00, V01, V10, V11, V20, V21, V30, V31);
        JROT(M00, M22, M02, M01, M12, M03, M23, V00, V02, V10, V12, V20, V22, V30, V32);
        JROT(M00, M33, M03, M01, M13, M02, M23, V00, V03, V10, V13, V20, V23, V30, V33);
        JROT(M11, M22, M12, M01, M02, M13, M23, V01, V02, V11, V12, V21, V22, V31, V32);
        JROT(M11, M33, M13, M01, M03, M12, M23, V01, V03, V11, V13, V21, V23, V31, V33);
        JROT(M22, M33, M23, M02, M03, M12, M13, V02, V03, V12, V13, V22, V23, V32, V33);
    }
#undef JROT

    // Column of the smallest eigenvalue.
    float ev0 = V00, ev1 = V10, ev2 = V20, ev3 = V30, mv = M00;
    if (M11 < mv) { mv = M11; ev0 = V01; ev1 = V11; ev2 = V21; ev3 = V31; }
    if (M22 < mv) { mv = M22; ev0 = V02; ev1 = V12; ev2 = V22; ev3 = V32; }
    if (M33 < mv) { mv = M33; ev0 = V03; ev1 = V13; ev2 = V23; ev3 = V33; }

    float denom = (fabsf(ev3) < 1e-8f) ? 1e-8f : ev3;
    float X0 = ev0 / denom, X1 = ev1 / denom, X2 = ev2 / denom;

    int   inlierNum = 0;
    float pconfSum = 0.f;
    float u0 = 0.f, v0 = 0.f;
    int   inl[S_FRAMES];
    for (int s = 0; s < S_FRAMES; ++s) {
        const float* P = w2cs + s * 16;
        float Xc0 = P[0] * X0 + P[1] * X1 + P[2]  * X2 + P[3];
        float Xc1 = P[4] * X0 + P[5] * X1 + P[6]  * X2 + P[7];
        float Xc2 = P[8] * X0 + P[9] * X1 + P[10] * X2 + P[11];
        float z = (fabsf(Xc2) < 1e-6f) ? 1e-6f : Xc2;
        float u = Xc0 / z, v = Xc1 / z;
        if (s == 0) { u0 = u; v0 = v; }
        float du = u - tnx[s], dv = v - tny[s];
        float err = sqrtf(du * du + dv * dv);
        int in = (err < 0.01f) ? 1 : 0;
        inl[s] = in;
        inlierNum += in;
    }
    for (int s = 0; s < S_FRAMES; ++s) pconfSum += tcf[s] * (float)inl[s];

    float fx0  = intr[0], fy0  = intr[4];
    float ppx0 = intr[2], ppy0 = intr[5];
    float px = u0 * fx0 + ppx0, py = v0 * fy0 + ppy0;
    float dx = px - xs0, dy = py - ys0;
    float errPx0 = sqrtf(dx * dx + dy * dy);
    int finalMask = ((errPx0 < 1.5f) && (inlierNum > S_FRAMES / 2)) ? 1 : 0;

    float pconf = pconfSum / (float)((inlierNum < 1) ? 1 : inlierNum);
    if (inlierNum < 4) pconf = 1.0f;   // MIN_TRI_INLIER

    out[t * 3 + 0] = X0;
    out[t * 3 + 1] = X1;
    out[t * 3 + 2] = X2;
    out[N_TRACKS * 3 + t] = pconf;
    out[N_TRACKS * 4 + t] = (float)finalMask;
    out[N_TRACKS * 5 + t] = (float)inlierNum;
}

// ---------------------------------------------------------------------------
// Launch
// ---------------------------------------------------------------------------
extern "C" void kernel_launch(void* const* d_in, const int* in_sizes, int n_in,
                              void* d_out, int out_size, void* d_ws, size_t ws_size,
                              hipStream_t stream) {
    const float* pts  = (const float*)d_in[0];   // (8,224,224,3)
    const float* conf = (const float*)d_in[1];   // (8,224,224)
    const float* intr = (const float*)d_in[2];   // (8,3,3)
    const float* w2cs = (const float*)d_in[3];   // (8,4,4)
    float* out = (float*)d_out;

    char* ws = (char*)d_ws;
    unsigned int* scal = (unsigned int*)ws;                        // 32 B scalars
    int*    idx0 = (int*)(ws + 32);                                // 8 KB
    float4* ref4 = (float4*)(ws + 32 + 8192);                      // 32 KB (16B aligned)
    unsigned long long* keys =
        (unsigned long long*)(ws + 32 + 8192 + 32768);             // 112 KB: [7][2048]
    const int nKeys = (S_FRAMES - 1) * N_TRACKS;

    topk_threshold_kernel<<<1, 1024, 0, stream>>>(conf, scal);
    topk_collect_kernel<<<(HW + 255) / 256, 256, 0, stream>>>(conf, scal, idx0);
    sort_gather_kernel<<<1, 1024, 0, stream>>>(conf, pts, idx0, ref4);
    init_keys_kernel<<<(nKeys + 255) / 256, 256, 0, stream>>>(keys, nKeys);

    dim3 mg(N_TRACKS / 16, S_FRAMES - 1, NSEG);
    match_kernel<<<mg, 32, 0, stream>>>(pts, ref4, keys);

    solve_kernel<<<(N_TRACKS + 255) / 256, 256, 0, stream>>>(conf, intr, w2cs, idx0, keys, out);
}
